// DotProductGraphAttention_47528108097841
// MI455X (gfx1250) — matrix operand order repaired and checked by reading, not compiled, not run on hardware
//
#include <hip/hip_runtime.h>
#include <math.h>

// MI455X / gfx1250 flash-attention for DotProductGraphAttention.
// B=8,H=8,N=1024,d=128 fp32. Output reshape is a raw view -> write [B,H,N,d] flat.
// Compute-bound (AI ~245 FLOP/B) -> use v_wmma_f32_16x16x32_f16 (f16 in, f32 acc).

typedef __attribute__((ext_vector_type(16))) _Float16 v16h;
typedef __attribute__((ext_vector_type(8)))  float    v8f;

#define WMMA16x16x32(A, B, C) \
  __builtin_amdgcn_wmma_f32_16x16x32_f16(false, (A), false, (B), (short)0, (C), false, false)

constexpr int   NN   = 1024;
constexpr int   DD   = 128;
constexpr float NEGC = -9e15f;

__global__ __launch_bounds__(128)
void graph_attn_fa(const float* __restrict__ Q, const float* __restrict__ K,
                   const float* __restrict__ V, const int* __restrict__ adj,
                   float* __restrict__ Out)
{
  __shared__ float pbuf[4][16 * 32];   // per-wave P-transpose staging (2 KB each)

  const int lane = threadIdx.x & 31;
  const int wave = threadIdx.x >> 5;
  const int head = blockIdx.x >> 4;          // b*H + h  (0..63)
  const int tb   = blockIdx.x & 15;          // 64-row block within head
  const int row0 = tb * 64 + wave * 16;      // first query row of this wave's tile
  const int col  = lane & 15;
  const int hi   = lane >> 4;                // lane half (0/1)
  const int h8   = hi * 8;
  const int coff = hi * 16;

  const float* Qh = Q + (size_t)head * NN * DD;
  const float* Kh = K + (size_t)head * NN * DD;
  const float* Vh = V + (size_t)head * NN * DD;
  float*       Oh = Out + (size_t)head * NN * DD;

  // ---- Q tile (16x128) as 4 f16 A-fragments (contiguous k order: k = 16*hi + s) ----
  v16h qa[4];
  {
    const float* qrow = Qh + (size_t)(row0 + col) * DD;
    #pragma unroll
    for (int c = 0; c < 4; ++c) {
      const float* p0 = qrow + c * 32 + coff;      // K = coff + 0..7
      const float* p1 = qrow + c * 32 + coff + 8;  // K = coff + 8 + 0..7
      v16h a;
      #pragma unroll
      for (int i = 0; i < 8; ++i) {
        a[i]     = (_Float16)p0[i];
        a[i + 8] = (_Float16)p1[i];
      }
      qa[c] = a;
    }
  }

  v8f acc[8] = {};                       // O accumulators: 8 col-tiles of 16x16 f32
  float mrow[8], lrow[8];
  #pragma unroll
  for (int r = 0; r < 8; ++r) { mrow[r] = -INFINITY; lrow[r] = 0.f; }

  for (int j0 = 0; j0 < NN; j0 += 32) {
    // ---- S = (Q K^T)/8 for 32 keys: two 16x16 C-tiles ----
    v8f s0 = {}, s1 = {};
    #pragma unroll
    for (int c = 0; c < 4; ++c) {
      // B fragment: lane = key column, halves = contraction K = coff + 0..15
      const float* k0p = Kh + (size_t)(j0 + col)      * DD + c * 32 + coff;
      const float* k1p = Kh + (size_t)(j0 + 16 + col) * DD + c * 32 + coff;
      v16h b0, b1;
      #pragma unroll
      for (int i = 0; i < 16; ++i) {
        b0[i] = (_Float16)k0p[i];
        b1[i] = (_Float16)k1p[i];
      }
      s0 = WMMA16x16x32(qa[c], b0, s0);
      s1 = WMMA16x16x32(qa[c], b1, s1);
    }

    // ---- mask + online softmax (row m = r + 8*hi, col = lane&15) ----
    float p0[8], p1[8], scl[8];
    #pragma unroll
    for (int r = 0; r < 8; ++r) {
      const int qabs = row0 + r + 8 * hi;
      float e0 = s0[r] * 0.125f;
      float e1 = s1[r] * 0.125f;
      e0 = (adj[(size_t)qabs * NN + j0 + col]      > 0) ? e0 : NEGC;
      e1 = (adj[(size_t)qabs * NN + j0 + 16 + col] > 0) ? e1 : NEGC;

      float tmx = fmaxf(e0, e1);
      tmx = fmaxf(tmx, __shfl_xor(tmx, 1));
      tmx = fmaxf(tmx, __shfl_xor(tmx, 2));
      tmx = fmaxf(tmx, __shfl_xor(tmx, 4));
      tmx = fmaxf(tmx, __shfl_xor(tmx, 8));
      const float mnew = fmaxf(mrow[r], tmx);
      scl[r]  = __expf(mrow[r] - mnew);   // exp(-inf)=0 on first block
      mrow[r] = mnew;
      p0[r] = __expf(e0 - mnew);          // NEGC-mnew -> 0; all-masked rows -> 1 (matches ref)
      p1[r] = __expf(e1 - mnew);
      float u = p0[r] + p1[r];
      u += __shfl_xor(u, 1);
      u += __shfl_xor(u, 2);
      u += __shfl_xor(u, 4);
      u += __shfl_xor(u, 8);
      lrow[r] = lrow[r] * scl[r] + u;
    }

    // ---- rescale accumulators (row mapping identical to C layout) ----
    #pragma unroll
    for (int t = 0; t < 8; ++t) {
      #pragma unroll
      for (int r = 0; r < 8; ++r) acc[t][r] *= scl[r];
    }

    // ---- transpose P (C layout -> A layout) through wave-private LDS ----
    float* pb = pbuf[wave];
    #pragma unroll
    for (int r = 0; r < 8; ++r) {
      const int m = r + 8 * hi;
      pb[m * 32 + col]      = p0[r];
      pb[m * 32 + 16 + col] = p1[r];
    }
    asm volatile("s_wait_dscnt 0x0" ::: "memory");  // DS in-order per wave; make visibility explicit
    v16h pa;
    {
      const float* pr = pb + col * 32;   // A lane = query row
      #pragma unroll
      for (int i = 0; i < 8; ++i) {
        pa[i]     = (_Float16)pr[coff + i];
        pa[i + 8] = (_Float16)pr[coff + 8 + i];
      }
    }

    // ---- O += P * V  (B fragment: lane = output dim, halves = key k = coff+i) ----
    #pragma unroll
    for (int t = 0; t < 8; ++t) {
      const float* vp = Vh + (size_t)(j0 + coff) * DD + t * 16 + col;
      v16h vb;
      #pragma unroll
      for (int i = 0; i < 16; ++i) vb[i] = (_Float16)vp[(size_t)i * DD];
      acc[t] = WMMA16x16x32(pa, vb, acc[t]);
    }
  }

  // ---- epilogue: divide by softmax denominator, store [B,H,N,d] flat ----
  #pragma unroll
  for (int r = 0; r < 8; ++r) {
    const int m = r + 8 * hi;
    const float inv = 1.f / lrow[r];
    float* orow = Oh + (size_t)(row0 + m) * DD + col;
    #pragma unroll
    for (int t = 0; t < 8; ++t) orow[t * 16] = acc[t][r] * inv;
  }
}

extern "C" void kernel_launch(void* const* d_in, const int* in_sizes, int n_in,
                              void* d_out, int out_size, void* d_ws, size_t ws_size,
                              hipStream_t stream) {
  const float* Q   = (const float*)d_in[0];
  const float* K   = (const float*)d_in[1];
  const float* V   = (const float*)d_in[2];
  const int*   adj = (const int*)d_in[3];
  float*       out = (float*)d_out;
  (void)in_sizes; (void)n_in; (void)out_size; (void)d_ws; (void)ws_size;

  // 64 heads * 16 row-blocks of 64; 128 threads = 4 wave32, one 16-row tile each
  dim3 grid(1024), block(128);
  hipLaunchKernelGGL(graph_attn_fa, grid, block, 0, stream, Q, K, V, adj, out);
}
